// MTM_29523605192753
// MI455X (gfx1250) — matrix operand order repaired and checked by reading, not compiled, not run on hardware
//
#include <hip/hip_runtime.h>
#include <hip/hip_bf16.h>

// Problem constants (match reference setup_inputs)
#define BB 8
#define TTT 4096
#define DDD 512
#define HHH 8
#define D3  1536
#define NF  576        // 9*C

typedef __attribute__((ext_vector_type(16))) __bf16 v16bf;
typedef __attribute__((ext_vector_type(8)))  float  v8f;

// A/B fragment for v_wmma_f32_16x16x32_bf16 (16 bf16 per lane)
union Frag16 { v16bf v; unsigned short u[16]; uint4 q[2]; };

__device__ __forceinline__ unsigned short f2bf(float f) {
  union { float f; unsigned int u; } x; x.f = f;
  unsigned int r = x.u + 0x7FFFu + ((x.u >> 16) & 1u);   // round-to-nearest-even
  return (unsigned short)(r >> 16);
}

__device__ __forceinline__ v8f wmma_bf16(const Frag16& a, const Frag16& b, v8f c) {
  // (neg_a, A, neg_b, B, c_mod, C, reuse_a, reuse_b)
  return __builtin_amdgcn_wmma_f32_16x16x32_bf16(false, a.v, false, b.v, (short)0, c,
                                                 false, false);
}

// ---------------------------------------------------------------------------
// 1) x (f32) -> bf16, vectorized
__global__ void k_cvt_x(const float* __restrict__ x, unsigned short* __restrict__ xb,
                        int n4) {
  int i = blockIdx.x * blockDim.x + threadIdx.x;
  if (i >= n4) return;
  const float4 f = ((const float4*)x)[i];
  uint2 o;
  o.x = (unsigned)f2bf(f.x) | ((unsigned)f2bf(f.y) << 16);
  o.y = (unsigned)f2bf(f.z) | ((unsigned)f2bf(f.w) << 16);
  ((uint2*)xb)[i] = o;
}

// 2) W_qkv (K=512, N=1536) f32 -> bf16 transposed to (N, K) for contiguous B-frags
__global__ void k_cvt_w(const float* __restrict__ w, unsigned short* __restrict__ wt) {
  int i = blockIdx.x * blockDim.x + threadIdx.x;       // i = n*512 + k
  if (i >= D3 * DDD) return;
  int n = i >> 9, k = i & 511;
  wt[i] = f2bf(w[(size_t)k * D3 + n]);
}

// ---------------------------------------------------------------------------
// 3) qkv = x @ W_qkv :  M=32768, K=512, N=1536, bf16 WMMA, 32x32 tile per wave
__global__ __launch_bounds__(256) void k_gemm_qkv(const unsigned short* __restrict__ Xb,
                                                  const unsigned short* __restrict__ Wt,
                                                  float* __restrict__ Y) {
  const int K = DDD;
  int wid  = (blockIdx.x * 256 + (int)threadIdx.x) >> 5;  // 1024*48 wave tiles
  int lane = threadIdx.x & 31;
  int tM = wid / 48, tN = wid % 48;
  int row0 = tM * 32, col0 = tN * 32;
  int m = lane & 15, hi = lane >> 4;
  v8f acc[2][2];
  for (int i = 0; i < 2; ++i)
    for (int j = 0; j < 2; ++j)
      for (int v = 0; v < 8; ++v) acc[i][j][v] = 0.f;
  for (int kk = 0; kk < K; kk += 32) {
    Frag16 a[2], b[2];
#pragma unroll
    for (int i = 0; i < 2; ++i) {   // A: elems 0..7 -> K=kk+hi*8.., 8..15 -> K=kk+16+hi*8..
      const unsigned short* p = Xb + (size_t)(row0 + i * 16 + m) * K + kk + hi * 8;
      a[i].q[0] = *(const uint4*)p;
      a[i].q[1] = *(const uint4*)(p + 16);
    }
#pragma unroll
    for (int j = 0; j < 2; ++j) {   // B: elems 0..7 -> K=kk+hi*8.., 8..15 -> K=kk+16+hi*8..
      const unsigned short* p = Wt + (size_t)(col0 + j * 16 + m) * K + kk + hi * 8;
      b[j].q[0] = *(const uint4*)p;
      b[j].q[1] = *(const uint4*)(p + 16);
    }
#pragma unroll
    for (int i = 0; i < 2; ++i)
#pragma unroll
      for (int j = 0; j < 2; ++j) acc[i][j] = wmma_bf16(a[i], b[j], acc[i][j]);
  }
#pragma unroll
  for (int i = 0; i < 2; ++i)
#pragma unroll
    for (int j = 0; j < 2; ++j) {
      float* o = Y + (size_t)(row0 + i * 16 + 8 * hi) * D3 + col0 + j * 16 + m;
#pragma unroll
      for (int v = 0; v < 8; ++v) o[(size_t)v * D3] = acc[i][j][v];
    }
}

// ---------------------------------------------------------------------------
// 4) depthwise k=3 conv over T on qkv (3D channels), float4 vectorized
__global__ void k_dwconv(const float* __restrict__ qkv, const float* __restrict__ wdw,
                         float* __restrict__ out) {
  int i = blockIdx.x * blockDim.x + threadIdx.x;     // (b, t, c4) ; c4 in [0,384)
  int c4 = i % (D3 / 4);
  int t  = (i / (D3 / 4)) % TTT;
  int b  = i / ((D3 / 4) * TTT);
  float4 o = {0.f, 0.f, 0.f, 0.f};
#pragma unroll
  for (int k = 0; k < 3; ++k) {
    int ts = t + k - 1;
    if (ts < 0 || ts >= TTT) continue;
    const float4 v = *(const float4*)(qkv + ((size_t)(b * TTT + ts)) * D3 + c4 * 4);
    o.x += v.x * wdw[(c4 * 4 + 0) * 3 + k];
    o.y += v.y * wdw[(c4 * 4 + 1) * 3 + k];
    o.z += v.z * wdw[(c4 * 4 + 2) * 3 + k];
    o.w += v.w * wdw[(c4 * 4 + 3) * 3 + k];
  }
  *(float4*)(out + ((size_t)(b * TTT + t)) * D3 + c4 * 4) = o;
}

// ---------------------------------------------------------------------------
// 5) fc branch: f[b,t,n*64+c] = b_fc[n] + sum_hh qkvc[b,t,hh*64+c] * W_fc[hh,n]
__global__ __launch_bounds__(256) void k_fc(const float* __restrict__ qkvc,
                                            const float* __restrict__ Wfc,
                                            const float* __restrict__ bfc,
                                            float* __restrict__ f) {
  __shared__ float sW[216];
  __shared__ float sb[9];
  int tid = threadIdx.x;
  if (tid < 216) sW[tid] = Wfc[tid];
  if (tid < 9)   sb[tid] = bfc[tid];
  __syncthreads();
  int i  = blockIdx.x * 256 + tid;
  int c  = i & 63;
  int bt = i >> 6;
  float in[24];
#pragma unroll
  for (int hh = 0; hh < 24; ++hh) in[hh] = qkvc[(size_t)bt * D3 + hh * 64 + c];
#pragma unroll
  for (int n = 0; n < 9; ++n) {
    float acc = sb[n];
#pragma unroll
    for (int hh = 0; hh < 24; ++hh) acc += in[hh] * sW[hh * 9 + n];
    f[(size_t)bt * NF + n * 64 + c] = acc;
  }
}

// ---------------------------------------------------------------------------
// 6) ctx branch (depthwise k3 -> relu -> depthwise k5), writes d_out
__global__ void k_ctx(const float* __restrict__ x, const float* __restrict__ w1,
                      const float* __restrict__ w2, float* __restrict__ out) {
  int i = blockIdx.x * blockDim.x + threadIdx.x;     // (b*T+t)*512 + d
  int d = i & 511;
  int t = (i >> 9) & 4095;
  float o = 0.f;
#pragma unroll
  for (int u = 0; u < 5; ++u) {
    int tc = t + u - 2;
    if (tc < 0 || tc >= TTT) continue;               // ctx1 zero-padded
    float s = 0.f;
#pragma unroll
    for (int k = 0; k < 3; ++k) {
      int ts = tc + k - 1;
      if (ts < 0 || ts >= TTT) continue;             // x zero-padded
      s += x[i + (ts - t) * DDD] * w1[d * 3 + k];
    }
    o += fmaxf(s, 0.f) * w2[d * 5 + u];
  }
  out[i] = o;
}

// 7) grouped conv w_dep over f (groups=C=64, 9 in / 8 out per group), adds to d_out
__global__ void k_convdep(const float* __restrict__ f, const float* __restrict__ wdep,
                          const float* __restrict__ bdep, float* __restrict__ out) {
  int i = blockIdx.x * blockDim.x + threadIdx.x;
  int d = i & 511;
  int t = (i >> 9) & 4095;
  int b = i >> 21;
  int g = d >> 3;                                    // 8 out channels per group
  float acc = bdep[d];
#pragma unroll
  for (int k = 0; k < 3; ++k) {
    int ts = t + k - 1;
    if (ts < 0 || ts >= TTT) continue;
    const float* fp = f + ((size_t)(b * TTT + ts)) * NF + 9 * g;
#pragma unroll
    for (int j = 0; j < 9; ++j) acc += wdep[d * 27 + j * 3 + k] * fp[j];
  }
  out[i] += acc;
}

// ---------------------------------------------------------------------------
// 8) inverse L2 norms over T for q (ch 0..511) and k (ch 512..1023)
__global__ __launch_bounds__(256) void k_norms(const float* __restrict__ qkvc,
                                               float* __restrict__ invn) {
  int wid  = (blockIdx.x * 256 + (int)threadIdx.x) >> 5;  // (b, ch) : 8*1024 waves
  int lane = threadIdx.x & 31;
  int b = wid >> 10, ch = wid & 1023;
  float s = 0.f;
  for (int t = lane; t < TTT; t += 32) {
    float v = qkvc[((size_t)(b * TTT + t)) * D3 + ch];
    s += v * v;
  }
#pragma unroll
  for (int o = 16; o > 0; o >>= 1) s += __shfl_xor(s, o, 32);
  if (lane == 0) invn[b * 1024 + ch] = 1.f / fmaxf(sqrtf(s), 1e-12f);
}

// 9) attn logits: per (b,h), 64x64 = q_n . k_n^T over T=4096, LDS-staged bf16 WMMA
__global__ __launch_bounds__(256) void k_attn_qk(const float* __restrict__ qkvc,
                                                 const float* __restrict__ invn,
                                                 const float* __restrict__ temp,
                                                 float* __restrict__ logits) {
  __shared__ unsigned short qs[32 * 64];   // (t, c) bf16, pre-normalized
  __shared__ unsigned short ks[32 * 64];
  int bh = blockIdx.x, b = bh >> 3, h = bh & 7;
  int tid = threadIdx.x, lane = tid & 31, w = tid >> 5;
  int m = lane & 15, hi = lane >> 4;
  int ti = w >> 1, tj0 = (2 * w) & 3, tj1 = tj0 + 1;   // 2 of 16 16x16 tiles per wave
  v8f acc0, acc1;
  for (int v = 0; v < 8; ++v) { acc0[v] = 0.f; acc1[v] = 0.f; }
  for (int t0 = 0; t0 < TTT; t0 += 32) {
#pragma unroll
    for (int e = tid; e < 2048; e += 256) {            // coalesced 64-wide loads
      int t = e >> 6, c = e & 63;
      size_t g = ((size_t)(b * TTT + t0 + t)) * D3 + h * 64 + c;
      qs[e] = f2bf(qkvc[g]       * invn[b * 1024 + h * 64 + c]);
      ks[e] = f2bf(qkvc[g + 512] * invn[b * 1024 + 512 + h * 64 + c]);
    }
    __syncthreads();
    Frag16 a, fb0, fb1;
#pragma unroll
    for (int e = 0; e < 8; ++e) {
      a.u[e]     = qs[(hi * 8 + e) * 64 + ti * 16 + m];
      a.u[8 + e] = qs[(16 + hi * 8 + e) * 64 + ti * 16 + m];
    }
#pragma unroll
    for (int e = 0; e < 16; ++e) {
      fb0.u[e] = ks[(hi * 8 + e + (e & 8)) * 64 + tj0 * 16 + m];
      fb1.u[e] = ks[(hi * 8 + e + (e & 8)) * 64 + tj1 * 16 + m];
    }
    acc0 = wmma_bf16(a, fb0, acc0);
    acc1 = wmma_bf16(a, fb1, acc1);
    __syncthreads();
  }
  float tp = temp[h];
#pragma unroll
  for (int v = 0; v < 8; ++v) {
    int c = ti * 16 + v + 8 * hi;
    logits[((size_t)bh * 64 + c) * 64 + tj0 * 16 + m] = acc0[v] * tp;
    logits[((size_t)bh * 64 + c) * 64 + tj1 * 16 + m] = acc1[v] * tp;
  }
}

// 10) softmax over last dim (64), one wave per row, output bf16
__global__ __launch_bounds__(256) void k_softmax(const float* __restrict__ logits,
                                                 unsigned short* __restrict__ attnb) {
  int row  = (blockIdx.x * 256 + (int)threadIdx.x) >> 5;  // B*H*C rows
  int lane = threadIdx.x & 31;
  size_t base = (size_t)row * 64;
  float x0 = logits[base + lane], x1 = logits[base + 32 + lane];
  float mx = fmaxf(x0, x1);
#pragma unroll
  for (int o = 16; o > 0; o >>= 1) mx = fmaxf(mx, __shfl_xor(mx, o, 32));
  float e0 = __expf(x0 - mx), e1 = __expf(x1 - mx);
  float s = e0 + e1;
#pragma unroll
  for (int o = 16; o > 0; o >>= 1) s += __shfl_xor(s, o, 32);
  float inv = 1.f / s;
  attnb[base + lane]      = f2bf(e0 * inv);
  attnb[base + 32 + lane] = f2bf(e1 * inv);
}

// 11) out_att = attn @ v : per (b,h) M=64(c), N=4096(t), K=64(d); adds to d_out
__global__ __launch_bounds__(256) void k_gemm_av(const unsigned short* __restrict__ attnb,
                                                 const float* __restrict__ qkvc,
                                                 float* __restrict__ out) {
  int wid  = (blockIdx.x * 256 + (int)threadIdx.x) >> 5;  // 64 bh * 256 tiles
  int lane = threadIdx.x & 31;
  int bh = wid >> 8, tl = wid & 255;
  int tM = tl >> 7, tN = tl & 127;
  int b = bh >> 3, h = bh & 7;
  int c0 = tM * 32, t0 = tN * 32;
  int m = lane & 15, hi = lane >> 4;
  v8f acc[2][2];
  for (int i = 0; i < 2; ++i)
    for (int j = 0; j < 2; ++j)
      for (int v = 0; v < 8; ++v) acc[i][j][v] = 0.f;
#pragma unroll
  for (int kk = 0; kk < 64; kk += 32) {
    Frag16 a[2], fb[2];
#pragma unroll
    for (int i = 0; i < 2; ++i) {   // attn row-major 64x64: contiguous K
      const unsigned short* p =
          attnb + ((size_t)bh * 64 + c0 + i * 16 + m) * 64 + kk + hi * 8;
      a[i].q[0] = *(const uint4*)p;
      a[i].q[1] = *(const uint4*)(p + 16);
    }
#pragma unroll
    for (int j = 0; j < 2; ++j) {   // B[d][t] = v[b,t,h*64+d] : d contiguous in memory
      const float* p = qkvc + ((size_t)(b * TTT + t0 + j * 16 + m)) * D3 +
                       1024 + h * 64 + kk + hi * 8;
#pragma unroll
      for (int e = 0; e < 16; ++e) fb[j].u[e] = f2bf(p[e + (e & 8)]);
    }
#pragma unroll
    for (int i = 0; i < 2; ++i)
#pragma unroll
      for (int j = 0; j < 2; ++j) acc[i][j] = wmma_bf16(a[i], fb[j], acc[i][j]);
  }
#pragma unroll
  for (int i = 0; i < 2; ++i)
#pragma unroll
    for (int j = 0; j < 2; ++j) {
      float* o = out + ((size_t)(b * TTT + t0 + j * 16 + m)) * DDD +
                 h * 64 + c0 + i * 16 + 8 * hi;     // 8 contiguous c per lane
#pragma unroll
      for (int v = 0; v < 8; ++v) o[v] += acc[i][j][v];
    }
}

// ---------------------------------------------------------------------------
extern "C" void kernel_launch(void* const* d_in, const int* in_sizes, int n_in,
                              void* d_out, int out_size, void* d_ws, size_t ws_size,
                              hipStream_t stream) {
  const float* x      = (const float*)d_in[0];
  const float* W_qkv  = (const float*)d_in[1];
  const float* w_dw   = (const float*)d_in[2];
  const float* w_ctx1 = (const float*)d_in[3];
  const float* w_ctx2 = (const float*)d_in[4];
  const float* W_fc   = (const float*)d_in[5];
  const float* b_fc   = (const float*)d_in[6];
  const float* w_dep  = (const float*)d_in[7];
  const float* b_dep  = (const float*)d_in[8];
  const float* temp   = (const float*)d_in[9];
  float* out = (float*)d_out;

  char* ws = (char*)d_ws;
  size_t cur = 0;
  auto take = [&](size_t bytes) -> char* {
    char* p = ws + cur;
    cur = (cur + bytes + 255) & ~(size_t)255;
    return p;
  };
  float*          qkv   = (float*)take((size_t)BB * TTT * D3 * 4);  // reused as f buffer
  float*          qkvc  = (float*)take((size_t)BB * TTT * D3 * 4);
  unsigned short* xb    = (unsigned short*)take((size_t)BB * TTT * DDD * 2);
  unsigned short* wt    = (unsigned short*)take((size_t)D3 * DDD * 2);
  float*          invn  = (float*)take((size_t)BB * 1024 * 4);
  float*          logit = (float*)take((size_t)BB * HHH * 64 * 64 * 4);
  unsigned short* attnb = (unsigned short*)take((size_t)BB * HHH * 64 * 64 * 2);
  float*          fbuf  = qkv;    // qkv dead after k_dwconv; f = 72MB fits in 192MB

  k_cvt_x  <<<(BB * TTT * DDD / 4 + 255) / 256, 256, 0, stream>>>(x, xb, BB * TTT * DDD / 4);
  k_cvt_w  <<<(D3 * DDD + 255) / 256,           256, 0, stream>>>(W_qkv, wt);
  k_gemm_qkv<<<(1024 * 48) / 8,                 256, 0, stream>>>(xb, wt, qkv);
  k_dwconv <<<BB * TTT * (D3 / 4) / 256,        256, 0, stream>>>(qkv, w_dw, qkvc);
  k_fc     <<<(BB * TTT * 64) / 256,            256, 0, stream>>>(qkvc, W_fc, b_fc, fbuf);
  k_ctx    <<<(BB * TTT * DDD) / 256,           256, 0, stream>>>(x, w_ctx1, w_ctx2, out);
  k_convdep<<<(BB * TTT * DDD) / 256,           256, 0, stream>>>(fbuf, w_dep, b_dep, out);
  k_norms  <<<(BB * 1024) / 8,                  256, 0, stream>>>(qkvc, invn);
  k_attn_qk<<<BB * HHH,                         256, 0, stream>>>(qkvc, invn, temp, logit);
  k_softmax<<<(BB * HHH * 64) / 8,              256, 0, stream>>>(logit, attnb);
  k_gemm_av<<<(BB * HHH * 256) / 8,             256, 0, stream>>>(attnb, qkvc, out);

  (void)in_sizes; (void)n_in; (void)out_size; (void)ws_size;
}
